// MECGraphNet_27693949125166
// MI455X (gfx1250) — compile-verified, matrix operand order checked by reading, not run on hardware
//
#include <hip/hip_runtime.h>
#include <hip/hip_bf16.h>
#include <math.h>

#define N_NODES 20000
#define N_EDGES 300000
#define HIDF    128
#define NLAY    4

typedef __attribute__((ext_vector_type(16))) _Float16 v16h;
typedef __attribute__((ext_vector_type(8)))  _Float16 v8h;
typedef __attribute__((ext_vector_type(8)))  float    v8f;

__device__ __forceinline__ float lrelu(float v) { return v >= 0.f ? v : 0.2f * v; }

// (lane, half) -> k-within-32 mapping for CDNA5 16-bit A/B fragments:
// lanes 0-15: halves 0..7 -> K0..7, halves 8..15 -> K16..23
// lanes 16-31: halves 0..7 -> K8..15, halves 8..15 -> K24..31
__device__ __forceinline__ int klocal(int lane, int h) {
  return (h & 7) + ((h & 8) << 1) + ((lane & 16) >> 1);
}

// Build one 16x32 f16 A-fragment from a packed-f16 row (32-half window p):
// two 16B loads, concatenated in registers (no converts).
__device__ __forceinline__ v16h load_a_chunk_h(const _Float16* p, int lane) {
  const _Float16* b = p + ((lane & 16) ? 8 : 0);
  v8h lo = *(const v8h*)(b);
  v8h hi = *(const v8h*)(b + 16);
  return __builtin_shufflevector(lo, hi, 0, 1, 2, 3, 4, 5, 6, 7,
                                 8, 9, 10, 11, 12, 13, 14, 15);
}

// ---------------- weight pre-swizzle into B-fragment order -------------------
// w1f layout: [layer][kc(8)][nt(8)][lane(32)][half(16)]  (ec_w1, K=256)
// w2f/gwf   : [layer][kc(4)][nt(8)][lane(32)][half(16)]  (ec_w2/gat_w, K=128)
__global__ __launch_bounds__(256) void prep_weights_kernel(
    const float* __restrict__ w1, const float* __restrict__ w2,
    const float* __restrict__ gw, _Float16* __restrict__ w1f,
    _Float16* __restrict__ w2f, _Float16* __restrict__ gwf) {
  int tid = blockIdx.x * 256 + threadIdx.x;   // 0..131071
  int h  = tid & 15;
  int l  = (tid >> 4) & 31;
  int nt = (tid >> 9) & 7;
  int kc = (tid >> 12) & 7;
  int i  = tid >> 15;
  int k = kc * 32 + klocal(l, h);
  int n = nt * 16 + (l & 15);
  w1f[tid] = (_Float16)w1[(size_t)(i * 256 + k) * HIDF + n];
  if (kc < 4) {
    int o2 = ((i * 4 + kc) * 8 + nt) * 512 + l * 16 + h;
    w2f[o2] = (_Float16)w2[(size_t)(i * HIDF + k) * HIDF + n];
    gwf[o2] = (_Float16)gw[(size_t)(i * HIDF + k) * HIDF + n];
  }
}

// ---------------- degree -----------------------------------------------------
__global__ void zero_deg_kernel(float* deg) {
  int i = blockIdx.x * 256 + threadIdx.x;
  if (i < N_NODES) deg[i] = 0.f;
}
__global__ void deg_kernel(const int* __restrict__ dst, float* deg) {
  int e = blockIdx.x * 256 + threadIdx.x;
  if (e < N_EDGES) atomicAdd(&deg[dst[e]], 1.f);
}

// ---------------- node encoder: xc = LN(lrelu(x@W+b)) ------------------------
__global__ __launch_bounds__(256) void encoder_kernel(
    const float* __restrict__ x, const float* __restrict__ w,
    const float* __restrict__ b, const float* __restrict__ g,
    const float* __restrict__ beta, float* __restrict__ xc,
    _Float16* __restrict__ xch) {
  int lane = threadIdx.x & 31;
  int node = blockIdx.x * 8 + (threadIdx.x >> 5);
  int n0 = lane * 4;
  float xf[4];
#pragma unroll
  for (int f = 0; f < 4; ++f) xf[f] = x[node * 4 + f];
  float vals[4], s1 = 0.f, s2 = 0.f;
#pragma unroll
  for (int j = 0; j < 4; ++j) {
    int n = n0 + j;
    float z = b[n];
#pragma unroll
    for (int f = 0; f < 4; ++f) z += xf[f] * w[f * HIDF + n];
    z = lrelu(z);
    vals[j] = z; s1 += z; s2 += z * z;
  }
#pragma unroll
  for (int o = 1; o < 32; o <<= 1) { s1 += __shfl_xor(s1, o, 32); s2 += __shfl_xor(s2, o, 32); }
  float mean = s1 / 128.f;
  float iv = rsqrtf(s2 / 128.f - mean * mean + 1e-5f);
#pragma unroll
  for (int j = 0; j < 4; ++j) {
    int n = n0 + j;
    float r = (vals[j] - mean) * iv * g[n] + beta[n];
    xc[(size_t)node * HIDF + n] = r;
    xch[(size_t)node * HIDF + n] = (_Float16)r;
  }
}

// ---------------- per-layer scratch init ------------------------------------
__global__ void layer_init_kernel(float* agg, float* emax, float* den,
                                  float* num, int useMax) {
  int i = blockIdx.x * 256 + threadIdx.x;          // grid == N*128 exactly
  agg[i] = useMax ? -3.0e38f : 0.f;
  num[i] = 0.f;
  if (i < N_NODES * 4) { emax[i] = -3.0e38f; den[i] = 0.f; }
}

// ---------------- fused EdgeConv: gather -> GEMM1 -> LN -> GEMM2 -> gate -> scatter
__global__ __launch_bounds__(64) void ec_edge_kernel(
    const _Float16* __restrict__ xch, const int* __restrict__ src,
    const int* __restrict__ dst, const _Float16* __restrict__ w1f,
    const float* __restrict__ b1, const float* __restrict__ lng,
    const float* __restrict__ lnb, const _Float16* __restrict__ w2f,
    const float* __restrict__ b2, const float* __restrict__ wa,
    const float* __restrict__ ba, float* __restrict__ agg, int useMax) {
  __shared__ _Float16 ldsh[2][16][136];   // 272B rows: 16B-aligned, conflict-free
  __shared__ float gateS[2][16];
  int lane = threadIdx.x & 31;
  int wave = threadIdx.x >> 5;
  int tile = blockIdx.x * 2 + wave;          // E/16 tiles, exact
  int eb = tile * 16;
  int m = lane & 15;
  int hb = (lane & 16) ? 8 : 0;
  int e = eb + m;
  int s = src[e], d = dst[e];
  const _Float16* rowD = xch + (size_t)d * HIDF;  // tmp[:,0:128]   = x_i (dst)
  const _Float16* rowS = xch + (size_t)s * HIDF;  // tmp[:,128:256] = x_j (src)

  v16h A[8];
#pragma unroll
  for (int kc = 0; kc < 4; ++kc) A[kc]     = load_a_chunk_h(rowD + kc * 32, lane);
#pragma unroll
  for (int kc = 0; kc < 4; ++kc) A[4 + kc] = load_a_chunk_h(rowS + kc * 32, lane);

  // edge-attention gate: sigmoid(tmp . wa + ba), row split across lane pairs
  float gp = 0.f;
#pragma unroll
  for (int kc = 0; kc < 8; ++kc)
#pragma unroll
    for (int h = 0; h < 16; ++h)
      gp += (float)A[kc][h] * wa[kc * 32 + klocal(lane, h)];
  gp += __shfl_xor(gp, 16, 32);
  float gate = 1.f / (1.f + __expf(-(gp + ba[0])));
  if (lane < 16) gateS[wave][m] = gate;

  // GEMM1: [16x256] x [256x128], K=256 (8 chunks), N=128 (8 tiles)
  v8f C[8];
#pragma unroll
  for (int nt = 0; nt < 8; ++nt) {
    float bias = b1[nt * 16 + m];
    v8f c;
#pragma unroll
    for (int r = 0; r < 8; ++r) c[r] = bias;
#pragma unroll
    for (int kc = 0; kc < 8; ++kc) {
      v16h B = *(const v16h*)(w1f + (size_t)(kc * 8 + nt) * 512 + lane * 16);
      c = __builtin_amdgcn_wmma_f32_16x16x32_f16(false, A[kc], false, B,
                                                 (short)0, c, false, false);
    }
#pragma unroll
    for (int r = 0; r < 8; ++r) c[r] = lrelu(c[r]);
    C[nt] = c;
  }

  // per-edge-row LayerNorm across 128 features (half-wave reductions)
  float mu[8], iv[8];
#pragma unroll
  for (int r = 0; r < 8; ++r) {
    float s1 = 0.f, s2 = 0.f;
#pragma unroll
    for (int nt = 0; nt < 8; ++nt) { float v = C[nt][r]; s1 += v; s2 += v * v; }
#pragma unroll
    for (int o = 1; o < 16; o <<= 1) { s1 += __shfl_xor(s1, o, 32); s2 += __shfl_xor(s2, o, 32); }
    float mean = s1 / 128.f;
    mu[r] = mean;
    iv[r] = rsqrtf(s2 / 128.f - mean * mean + 1e-5f);
  }
#pragma unroll
  for (int nt = 0; nt < 8; ++nt) {
    float gl = lng[nt * 16 + m], bl = lnb[nt * 16 + m];
#pragma unroll
    for (int r = 0; r < 8; ++r)
      ldsh[wave][r + hb][nt * 16 + m] =
          (_Float16)((C[nt][r] - mu[r]) * iv[r] * gl + bl);
  }
  __syncthreads();

  // re-fragment from LDS for GEMM2 (K=128, 4 chunks), direct f16 loads
  v16h A2[4];
#pragma unroll
  for (int kc = 0; kc < 4; ++kc)
    A2[kc] = load_a_chunk_h(&ldsh[wave][m][kc * 32], lane);

  float gr[8];
  int dnode[8];
#pragma unroll
  for (int r = 0; r < 8; ++r) {
    gr[r] = gateS[wave][r + hb];
    dnode[r] = dst[eb + r + hb];
  }

  // GEMM2 + lrelu + gate + scatter (atomic max / add into agg, L2-resident)
#pragma unroll
  for (int nt = 0; nt < 8; ++nt) {
    float bias = b2[nt * 16 + m];
    v8f c;
#pragma unroll
    for (int r = 0; r < 8; ++r) c[r] = bias;
#pragma unroll
    for (int kc = 0; kc < 4; ++kc) {
      v16h B = *(const v16h*)(w2f + (size_t)(kc * 8 + nt) * 512 + lane * 16);
      c = __builtin_amdgcn_wmma_f32_16x16x32_f16(false, A2[kc], false, B,
                                                 (short)0, c, false, false);
    }
#pragma unroll
    for (int r = 0; r < 8; ++r) {
      float v = lrelu(c[r]) * gr[r];
      float* ap = agg + (size_t)dnode[r] * HIDF + nt * 16 + m;
      if (useMax)
        __hip_atomic_fetch_max(ap, v, __ATOMIC_RELAXED, __HIP_MEMORY_SCOPE_AGENT);
      else
        atomicAdd(ap, v);
    }
  }
}

// ---------------- GAT: wx = xc @ gat_w via WMMA ------------------------------
__global__ __launch_bounds__(64) void gat_node_kernel(
    const _Float16* __restrict__ xch, const _Float16* __restrict__ gwf,
    float* __restrict__ wx) {
  int lane = threadIdx.x & 31;
  int wave = threadIdx.x >> 5;
  int tile = blockIdx.x * 2 + wave;          // N/16 tiles, exact
  int nb = tile * 16;
  int m = lane & 15;
  int hb = (lane & 16) ? 8 : 0;
  const _Float16* row = xch + (size_t)(nb + m) * HIDF;
  v16h A[4];
#pragma unroll
  for (int kc = 0; kc < 4; ++kc) A[kc] = load_a_chunk_h(row + kc * 32, lane);
#pragma unroll
  for (int nt = 0; nt < 8; ++nt) {
    v8f c;
#pragma unroll
    for (int r = 0; r < 8; ++r) c[r] = 0.f;
#pragma unroll
    for (int kc = 0; kc < 4; ++kc) {
      v16h B = *(const v16h*)(gwf + (size_t)(kc * 8 + nt) * 512 + lane * 16);
      c = __builtin_amdgcn_wmma_f32_16x16x32_f16(false, A[kc], false, B,
                                                 (short)0, c, false, false);
    }
#pragma unroll
    for (int r = 0; r < 8; ++r)
      wx[(size_t)(nb + r + hb) * HIDF + nt * 16 + m] = c[r];
  }
}

// asrc/adst = (wx.reshape(N,4,32) * a).sum(-1)  (wave per node)
__global__ __launch_bounds__(256) void gat_attn_kernel(
    const float* __restrict__ wx, const float* __restrict__ As,
    const float* __restrict__ Ad, float* __restrict__ asrcN,
    float* __restrict__ adstN) {
  int lane = threadIdx.x & 31;
  int node = blockIdx.x * 8 + (threadIdx.x >> 5);
  int n0 = lane * 4;
  int head = lane >> 3;
  float ps = 0.f, pd = 0.f;
#pragma unroll
  for (int j = 0; j < 4; ++j) {
    int n = n0 + j;
    float w = wx[(size_t)node * HIDF + n];
    ps += w * As[head * 32 + (n & 31)];
    pd += w * Ad[head * 32 + (n & 31)];
  }
#pragma unroll
  for (int o = 1; o < 8; o <<= 1) { ps += __shfl_xor(ps, o, 32); pd += __shfl_xor(pd, o, 32); }
  if ((lane & 7) == 0) {
    asrcN[node * 4 + head] = ps;
    adstN[node * 4 + head] = pd;
  }
}

// pass1: segment max of e = lrelu(asrc[src]+adst[dst]) over dst (incl. self loops)
__global__ void gat_e1_kernel(const int* __restrict__ src,
                              const int* __restrict__ dst,
                              const float* __restrict__ asrcN,
                              const float* __restrict__ adstN,
                              float* __restrict__ emax) {
  int idx = blockIdx.x * 256 + threadIdx.x;
  if (idx >= N_EDGES + N_NODES) return;
  int s, d;
  if (idx < N_EDGES) { s = src[idx]; d = dst[idx]; } else { s = d = idx - N_EDGES; }
#pragma unroll
  for (int h = 0; h < 4; ++h) {
    float e = lrelu(asrcN[s * 4 + h] + adstN[d * 4 + h]);
    __hip_atomic_fetch_max(&emax[d * 4 + h], e, __ATOMIC_RELAXED,
                           __HIP_MEMORY_SCOPE_AGENT);
  }
}

// pass2: p = exp(e - emax[dst]); den[dst] += p; num[dst] += p * wx[src]
__global__ __launch_bounds__(256) void gat_e2_kernel(
    const int* __restrict__ src, const int* __restrict__ dst,
    const float* __restrict__ asrcN, const float* __restrict__ adstN,
    const float* __restrict__ emax, const float* __restrict__ wx,
    float* __restrict__ den, float* __restrict__ num) {
  int lane = threadIdx.x & 31;
  int idx = blockIdx.x * 8 + (threadIdx.x >> 5);   // wave per edge, exact
  int s, d;
  if (idx < N_EDGES) { s = src[idx]; d = dst[idx]; } else { s = d = idx - N_EDGES; }
  int head = lane >> 3;
  int n0 = lane * 4;
  float e = lrelu(asrcN[s * 4 + head] + adstN[d * 4 + head]);
  float p = __expf(e - emax[d * 4 + head]);
  if ((lane & 7) == 0) atomicAdd(&den[d * 4 + head], p);
#pragma unroll
  for (int j = 0; j < 4; ++j) {
    int n = n0 + j;
    atomicAdd(&num[(size_t)d * HIDF + n], p * wx[(size_t)s * HIDF + n]);
  }
}

// xc = lrelu(LN(res + agg + num/den + gat_bias))   (wave per node, in place)
__global__ __launch_bounds__(256) void node_update_kernel(
    float* __restrict__ xc, _Float16* __restrict__ xch,
    const float* __restrict__ agg, const float* __restrict__ num,
    const float* __restrict__ den, const float* __restrict__ deg,
    const float* __restrict__ gbias, const float* __restrict__ lng,
    const float* __restrict__ lnb, int useMax) {
  int lane = threadIdx.x & 31;
  int node = blockIdx.x * 8 + (threadIdx.x >> 5);
  int n0 = lane * 4;
  float dg = deg[node];
  float dn = den[node * 4 + (lane >> 3)];
  float vals[4], s1 = 0.f, s2 = 0.f;
#pragma unroll
  for (int j = 0; j < 4; ++j) {
    int n = n0 + j;
    float a = agg[(size_t)node * HIDF + n];
    if (useMax) a = (dg > 0.f) ? a : 0.f;
    else        a = a / fmaxf(dg, 1.f);
    float g = num[(size_t)node * HIDF + n] / dn + gbias[n];
    float v = xc[(size_t)node * HIDF + n] + a + g;
    vals[j] = v; s1 += v; s2 += v * v;
  }
#pragma unroll
  for (int o = 1; o < 32; o <<= 1) { s1 += __shfl_xor(s1, o, 32); s2 += __shfl_xor(s2, o, 32); }
  float mean = s1 / 128.f;
  float iv = rsqrtf(s2 / 128.f - mean * mean + 1e-5f);
#pragma unroll
  for (int j = 0; j < 4; ++j) {
    int n = n0 + j;
    float r = lrelu((vals[j] - mean) * iv * lng[n] + lnb[n]);
    xc[(size_t)node * HIDF + n] = r;
    xch[(size_t)node * HIDF + n] = (_Float16)r;
  }
}

// ---------------- dueling heads on node 0 ------------------------------------
__global__ __launch_bounds__(128) void head_kernel(
    const float* __restrict__ xc,
    const float* aw1, const float* ab1, const float* ag, const float* abeta,
    const float* aw2, const float* ab2,
    const float* vw1, const float* vb1, const float* vg, const float* vbeta,
    const float* vw2, const float* vb2, float* __restrict__ out) {
  __shared__ float tS[128], z1S[128], z2S[128], aS[128], vS[128], qS[10], vOut;
  int n = threadIdx.x;
  tS[n] = xc[n];
  __syncthreads();
  // advantage branch
  float z = ab1[n];
  for (int k = 0; k < 128; ++k) z += tS[k] * aw1[k * HIDF + n];
  z = lrelu(z);
  z1S[n] = z;
  __syncthreads();
  float m = 0.f;
  for (int k = 0; k < 128; ++k) m += z1S[k];
  m /= 128.f;
  float v = 0.f;
  for (int k = 0; k < 128; ++k) { float dd = z1S[k] - m; v += dd * dd; }
  v /= 128.f;
  aS[n] = (z - m) * rsqrtf(v + 1e-5f) * ag[n] + abeta[n];
  // value branch
  float y = vb1[n];
  for (int k = 0; k < 128; ++k) y += tS[k] * vw1[k * HIDF + n];
  y = lrelu(y);
  z2S[n] = y;
  __syncthreads();
  m = 0.f;
  for (int k = 0; k < 128; ++k) m += z2S[k];
  m /= 128.f;
  v = 0.f;
  for (int k = 0; k < 128; ++k) { float dd = z2S[k] - m; v += dd * dd; }
  v /= 128.f;
  vS[n] = (y - m) * rsqrtf(v + 1e-5f) * vg[n] + vbeta[n];
  __syncthreads();
  if (n < 10) {
    float q = ab2[n];
    for (int k = 0; k < 128; ++k) q += aS[k] * aw2[k * 10 + n];
    qS[n] = q;
  }
  if (n == 127) {
    float vv = vb2[0];
    for (int k = 0; k < 128; ++k) vv += vS[k] * vw2[k];
    vOut = vv;
  }
  __syncthreads();
  if (n == 0) {
    float vv = vOut;
    float am = 0.f;
    for (int j = 0; j < 10; ++j) am += qS[j];
    am /= 10.f;
    for (int j = 0; j < 10; ++j) out[j] = vv + qS[j] - am;
    out[10] = vv;
  }
}

// ---------------- host driver ------------------------------------------------
extern "C" void kernel_launch(void* const* d_in, const int* in_sizes, int n_in,
                              void* d_out, int out_size, void* d_ws,
                              size_t ws_size, hipStream_t stream) {
  const float* x       = (const float*)d_in[0];
  const int*   ei      = (const int*)d_in[1];
  // d_in[2] edge_attr, d_in[7..10] edge encoder: computed but unused by output
  const float* ne_w    = (const float*)d_in[3];
  const float* ne_b    = (const float*)d_in[4];
  const float* ne_g    = (const float*)d_in[5];
  const float* ne_beta = (const float*)d_in[6];
  const float* ec_w1   = (const float*)d_in[11];
  const float* ec_b1   = (const float*)d_in[12];
  const float* ec_lng  = (const float*)d_in[13];
  const float* ec_lnb  = (const float*)d_in[14];
  const float* ec_w2   = (const float*)d_in[15];
  const float* ec_b2   = (const float*)d_in[16];
  const float* ec_wa   = (const float*)d_in[17];
  const float* ec_ba   = (const float*)d_in[18];
  const float* gat_w   = (const float*)d_in[19];
  const float* gat_as  = (const float*)d_in[20];
  const float* gat_ad  = (const float*)d_in[21];
  const float* gat_b   = (const float*)d_in[22];
  const float* ln_g    = (const float*)d_in[23];
  const float* ln_b    = (const float*)d_in[24];
  const float* adv_w1  = (const float*)d_in[25];
  const float* adv_b1  = (const float*)d_in[26];
  const float* adv_g   = (const float*)d_in[27];
  const float* adv_be  = (const float*)d_in[28];
  const float* adv_w2  = (const float*)d_in[29];
  const float* adv_b2  = (const float*)d_in[30];
  const float* val_w1  = (const float*)d_in[31];
  const float* val_b1  = (const float*)d_in[32];
  const float* val_g   = (const float*)d_in[33];
  const float* val_be  = (const float*)d_in[34];
  const float* val_w2  = (const float*)d_in[35];
  const float* val_b2  = (const float*)d_in[36];

  const int* src = ei;
  const int* dst = ei + N_EDGES;

  float* f    = (float*)d_ws;
  float* xc   = f;                          // N*128
  float* agg  = xc + (size_t)N_NODES * HIDF;
  float* wx   = agg + (size_t)N_NODES * HIDF;
  float* num  = wx + (size_t)N_NODES * HIDF;
  float* asrc = num + (size_t)N_NODES * HIDF;  // N*4
  float* adst = asrc + (size_t)N_NODES * 4;
  float* emax = adst + (size_t)N_NODES * 4;
  float* den  = emax + (size_t)N_NODES * 4;
  float* deg  = den + (size_t)N_NODES * 4;     // N
  _Float16* w1f = (_Float16*)(deg + N_NODES);  // NL*32768 halves
  _Float16* w2f = w1f + (size_t)NLAY * 32768;  // NL*16384
  _Float16* gwf = w2f + (size_t)NLAY * 16384;  // NL*16384
  _Float16* xch = gwf + (size_t)NLAY * 16384;  // N*128 halves

  prep_weights_kernel<<<512, 256, 0, stream>>>(ec_w1, ec_w2, gat_w, w1f, w2f, gwf);
  zero_deg_kernel<<<(N_NODES + 255) / 256, 256, 0, stream>>>(deg);
  deg_kernel<<<(N_EDGES + 255) / 256, 256, 0, stream>>>(dst, deg);
  encoder_kernel<<<N_NODES / 8, 256, 0, stream>>>(x, ne_w, ne_b, ne_g, ne_beta,
                                                  xc, xch);

  for (int i = 0; i < NLAY; ++i) {
    int useMax = (i % 2 == 0) ? 1 : 0;
    layer_init_kernel<<<(N_NODES * HIDF) / 256, 256, 0, stream>>>(agg, emax, den, num, useMax);
    ec_edge_kernel<<<N_EDGES / 32, 64, 0, stream>>>(
        xch, src, dst, w1f + (size_t)i * 32768, ec_b1 + i * HIDF,
        ec_lng + i * HIDF, ec_lnb + i * HIDF, w2f + (size_t)i * 16384,
        ec_b2 + i * HIDF, ec_wa + i * 256, ec_ba + i, agg, useMax);
    gat_node_kernel<<<N_NODES / 32, 64, 0, stream>>>(xch, gwf + (size_t)i * 16384, wx);
    gat_attn_kernel<<<N_NODES / 8, 256, 0, stream>>>(wx, gat_as + i * 128,
                                                     gat_ad + i * 128, asrc, adst);
    gat_e1_kernel<<<(N_EDGES + N_NODES + 255) / 256, 256, 0, stream>>>(src, dst, asrc, adst, emax);
    gat_e2_kernel<<<(N_EDGES + N_NODES) / 8, 256, 0, stream>>>(src, dst, asrc, adst,
                                                               emax, wx, den, num);
    node_update_kernel<<<N_NODES / 8, 256, 0, stream>>>(
        xc, xch, agg, num, den, deg, gat_b + i * HIDF, ln_g + i * HIDF,
        ln_b + i * HIDF, useMax);
  }

  head_kernel<<<1, 128, 0, stream>>>(xc, adv_w1, adv_b1, adv_g, adv_be, adv_w2,
                                     adv_b2, val_w1, val_b1, val_g, val_be,
                                     val_w2, val_b2, (float*)d_out);
}
